// LinearAttention_40252433498403
// MI455X (gfx1250) — matrix operand order repaired and checked by reading, not compiled, not run on hardware
//
#include <hip/hip_runtime.h>

#define DEVFN __device__ __forceinline__

typedef __attribute__((ext_vector_type(16))) __bf16 v16bf;
typedef __attribute__((ext_vector_type(8)))  float  v8f;
typedef __attribute__((ext_vector_type(4)))  unsigned int v4u;
typedef __attribute__((ext_vector_type(8)))  int v8i;
typedef __attribute__((ext_vector_type(4)))  int v4i;

union Frag { v16bf v; unsigned short u[16]; uint4 q[2]; };

static constexpr int BATCH = 16;
static constexpr int C     = 256;
static constexpr int N     = 4096;   // 64*64
static constexpr int HID   = 512;    // 8 heads * 64
static constexpr int HEADS = 8;
static constexpr int DH    = 64;
static constexpr int BH    = BATCH * HEADS;   // 128
static constexpr int K3    = 3 * HID;         // 1536

#if defined(__has_builtin)
#if __has_builtin(__builtin_amdgcn_tensor_load_to_lds) && __has_builtin(__builtin_amdgcn_s_wait_tensorcnt)
#define HAVE_TDM 1
#endif
#endif
#ifndef HAVE_TDM
#define HAVE_TDM 0
#endif

DEVFN unsigned short f2bf(float f) {
  unsigned u = __float_as_uint(f);
  u += 0x7fffu + ((u >> 16) & 1u);           // round-to-nearest-even
  return (unsigned short)(u >> 16);
}
DEVFN float bf2f(unsigned short h) { return __uint_as_float(((unsigned)h) << 16); }
DEVFN unsigned pack2(float a, float b) {
  return (unsigned)f2bf(a) | ((unsigned)f2bf(b) << 16);
}

#if HAVE_TDM
// TDM copy of `n2` contiguous 2-byte elements global -> LDS (one descriptor,
// 1-row tile). D# layout per CDNA5 ISA ch.8 (group0 128b, group1 256b).
DEVFN void tdm_copy_to_lds(unsigned lds_addr, const void* gptr, unsigned n2) {
  const unsigned long long ga = (unsigned long long)(size_t)gptr;
  v4u g0;
  g0.x = 1u;                                  // count=1, user mode, no gather
  g0.y = lds_addr;                            // lds_addr[31:0]
  g0.z = (unsigned)ga;                        // global_addr[31:0]
  g0.w = (unsigned)(ga >> 32) | (2u << 30);   // global_addr[56:32] | type=2
  v8i g1;
  g1[0] = 0x00010000;                         // wg_mask=0, data_size=1 (2B)
  g1[1] = (int)((n2 & 0xffffu) << 16);        // tensor_dim0[15:0]  @bits[63:48]
  g1[2] = (int)((n2 >> 16) & 0xffffu) | (1 << 16); // td0[31:16], tensor_dim1=1
  g1[3] = (int)((n2 & 0xffffu) << 16);        // tile_dim0 @bits[127:112]
  g1[4] = 1;                                  // tile_dim1=1, tile_dim2=0
  g1[5] = (int)n2;                            // tensor_dim0_stride[31:0]
  g1[6] = 0;                                  // stride hi / dim1_stride lo
  g1[7] = 0;
  const v4i z4 = {0, 0, 0, 0};
#if __clang_major__ >= 23
  const v8i z8 = {0, 0, 0, 0, 0, 0, 0, 0};
  __builtin_amdgcn_tensor_load_to_lds(g0, g1, z4, z4, z8, 0);
#else
  __builtin_amdgcn_tensor_load_to_lds(g0, g1, z4, z4, 0);
#endif
}
#endif

// ---------------------------------------------------------------- utilities
__global__ void cvt_f32_bf16(const float* __restrict__ in,
                             unsigned short* __restrict__ out, int n) {
  int i = blockIdx.x * 256 + threadIdx.x;
  if (i < n) out[i] = f2bf(in[i]);
}

// ------------------------------------------------- kernel 1: channel LN -> bf16
__global__ void ln_in(const float* __restrict__ f, const float* __restrict__ g,
                      unsigned short* __restrict__ X) {
  const int pix = blockIdx.x * 256 + threadIdx.x;        // 0 .. B*N-1
  const int b = pix >> 12;
  const int n = pix & (N - 1);
  const float* p = f + (size_t)b * C * N + n;
  float s = 0.f, ss = 0.f;
  for (int c = 0; c < C; ++c) { float v = p[(size_t)c * N]; s += v; ss += v * v; }
  const float mu   = s * (1.0f / C);
  const float var  = ss * (1.0f / C) - mu * mu;
  const float rstd = rsqrtf(var + 1e-5f);
  unsigned short* xo = X + (size_t)b * C * N + n;
  for (int c = 0; c < C; ++c)
    xo[(size_t)c * N] = f2bf((p[(size_t)c * N] - mu) * rstd * g[c]);
}

// --------------------------- kernel 2: QKV GEMM + fused q-softmax / v-norm
// grid (32, 24, 16), block 128 (4 waves). Wave: M=64 (one head) x N=32.
__global__ void qkv_gemm(const unsigned short* __restrict__ Wbf,
                         const unsigned short* __restrict__ X,
                         unsigned short* __restrict__ Q,
                         unsigned short* __restrict__ Kb,
                         unsigned short* __restrict__ V) {
  const int lane = threadIdx.x & 31;
  const int wave = threadIdx.x >> 5;
  const int b    = blockIdx.z;
  const int m0   = blockIdx.y * 64;
  const int n0   = blockIdx.x * 128 + wave * 32;
  const int ln   = lane & 15, half = lane >> 4;

  const unsigned short* Xb = X + (size_t)b * C * N;
  const v8f vz = {0.f,0.f,0.f,0.f,0.f,0.f,0.f,0.f};
  v8f acc[4][2];
  for (int i = 0; i < 4; ++i) for (int j = 0; j < 2; ++j) acc[i][j] = vz;

  for (int kk = 0; kk < C; kk += 32) {
    Frag a[4];
#pragma unroll
    for (int tm = 0; tm < 4; ++tm) {
      const uint4* p = (const uint4*)(Wbf + (size_t)(m0 + tm * 16 + ln) * C + kk + 8 * half);
      a[tm].q[0] = p[0];          // k = kk+8g .. +7
      a[tm].q[1] = p[2];          // k = kk+16+8g .. +7
    }
    Frag bb[2];
#pragma unroll
    for (int tn = 0; tn < 2; ++tn) {
      for (int i = 0; i < 16; ++i)
        bb[tn].u[i] = Xb[(size_t)(kk + ((i < 8) ? i + 8 * half : i + 8 + 8 * half)) * N + n0 + tn * 16 + ln];
    }
#pragma unroll
    for (int tm = 0; tm < 4; ++tm)
#pragma unroll
      for (int tn = 0; tn < 2; ++tn)
        acc[tm][tn] = __builtin_amdgcn_wmma_f32_16x16x32_bf16(
            false, a[tm].v, false, bb[tn].v, (short)0, acc[tm][tn], false, false);
  }

  const int region = m0 / HID;              // 0=q 1=k 2=v
  const int head   = (m0 - region * HID) / DH;
  const size_t bh  = (size_t)b * HEADS + head;
  unsigned short* outp = (region == 0) ? Q : (region == 1) ? Kb : V;

#pragma unroll
  for (int tn = 0; tn < 2; ++tn) {
    const int n = n0 + tn * 16 + ln;
    float scale = 1.0f, bias = 0.0f; bool do_exp = false;
    if (region == 0) {                       // softmax over d (wave-local: 4 tiles + xor16)
      float mx = -1e30f;
      for (int tm = 0; tm < 4; ++tm) for (int r = 0; r < 8; ++r) mx = fmaxf(mx, acc[tm][tn][r]);
      mx = fmaxf(mx, __shfl_xor(mx, 16, 32));
      float s = 0.f;
      for (int tm = 0; tm < 4; ++tm) for (int r = 0; r < 8; ++r) s += __expf(acc[tm][tn][r] - mx);
      s += __shfl_xor(s, 16, 32);
      do_exp = true; bias = mx; scale = 0.125f / s;        // * DIM_HEAD^-0.5
    } else if (region == 2) {                // L2 norm over d, * 1/sqrt(N)
      float ss = 0.f;
      for (int tm = 0; tm < 4; ++tm) for (int r = 0; r < 8; ++r) { float t = acc[tm][tn][r]; ss += t * t; }
      ss += __shfl_xor(ss, 16, 32);
      scale = (1.0f / 64.0f) / fmaxf(sqrtf(ss), 1e-12f);
    }
#pragma unroll
    for (int tm = 0; tm < 4; ++tm) {
      float vv[8];
      for (int r = 0; r < 8; ++r) {
        float t = acc[tm][tn][r];
        vv[r] = do_exp ? __expf(t - bias) * scale : t * scale;
      }
      uint4 st;
      st.x = pack2(vv[0], vv[1]); st.y = pack2(vv[2], vv[3]);
      st.z = pack2(vv[4], vv[5]); st.w = pack2(vv[6], vv[7]);
      *(uint4*)(outp + (bh * N + n) * (size_t)DH + tm * 16 + 8 * half) = st;
    }
  }
}

// ----------------------- kernel 3: k online-softmax column stats over N
__global__ void k_colstats(const unsigned short* __restrict__ Kb,
                           float2* __restrict__ stats) {
  __shared__ float smx[4][64];
  __shared__ float ssm[4][64];
  const int bh = blockIdx.x;
  const int d  = threadIdx.x & 63;
  const int g  = threadIdx.x >> 6;
  const unsigned short* p = Kb + (size_t)bh * N * DH + d;
  float m = -1e30f, s = 0.f;
  for (int n = g; n < N; n += 4) {
    float v = bf2f(p[(size_t)n * DH]);
    if (v > m) { s = s * __expf(m - v) + 1.0f; m = v; }
    else        s += __expf(v - m);
  }
  smx[g][d] = m; ssm[g][d] = s;
  __syncthreads();
  if (threadIdx.x < 64) {
    float M = smx[0][d];
    for (int i = 1; i < 4; ++i) M = fmaxf(M, smx[i][d]);
    float S = 0.f;
    for (int i = 0; i < 4; ++i) S += ssm[i][d] * __expf(smx[i][d] - M);
    stats[(size_t)bh * 64 + d] = make_float2(M, S);
  }
}

// ----------------------- kernel 4: normalize k in place (softmax / sqrt(N))
__global__ void k_norm(unsigned short* __restrict__ Kb,
                       const float2* __restrict__ stats) {
  const size_t total = (size_t)BH * N * DH;
  for (size_t i = blockIdx.x * 256ull + threadIdx.x; i < total;
       i += (size_t)gridDim.x * 256ull) {
    const int d = (int)(i & 63);
    const size_t bh = i >> 18;                      // / (N*DH)
    const float2 st = stats[bh * 64 + d];
    Kb[i] = f2bf(__expf(bf2f(Kb[i]) - st.x) / st.y * (1.0f / 64.0f));
  }
}

// ----------------------- kernel 5: context = k^T v  (per bh: 64x64, K=4096)
// Block 256 (8 waves) per bh. 64-row k'/v tiles staged to LDS via the Tensor
// Data Mover (one wave issues, s_wait_tensorcnt + barrier publishes), then the
// k^T gather becomes cheap LDS traffic instead of strided global u16 loads.
__global__ void ctx_gemm(const unsigned short* __restrict__ Kb,
                         const unsigned short* __restrict__ V,
                         unsigned short* __restrict__ CTX) {
  __shared__ unsigned short lk[64 * DH];   // 8 KiB
  __shared__ unsigned short lv[64 * DH];   // 8 KiB
  const int lane = threadIdx.x & 31;
  const int wave = threadIdx.x >> 5;
  const int bh   = blockIdx.x;
  const int tm   = wave & 3;
  const int tn0  = (wave >> 2) * 2;
  const int ln   = lane & 15, half = lane >> 4;
  const unsigned short* Kp = Kb + (size_t)bh * N * DH;
  const unsigned short* Vp = V  + (size_t)bh * N * DH;
  const v8f vz = {0.f,0.f,0.f,0.f,0.f,0.f,0.f,0.f};
  v8f acc[2] = {vz, vz};
  const int m = tm * 16 + ln;                       // d row of A = k^T

  for (int n0 = 0; n0 < N; n0 += 64) {
    __syncthreads();                                // previous tile fully consumed
#if HAVE_TDM
    if (threadIdx.x < 32) {                         // wave 0 drives the TDM
      tdm_copy_to_lds((unsigned)(size_t)(void*)lk, Kp + (size_t)n0 * DH, 64 * DH);
      tdm_copy_to_lds((unsigned)(size_t)(void*)lv, Vp + (size_t)n0 * DH, 64 * DH);
      __builtin_amdgcn_s_wait_tensorcnt(0);
    }
#else
    for (int t = threadIdx.x; t < 64 * DH / 8; t += 256) {
      ((uint4*)lk)[t] = ((const uint4*)(Kp + (size_t)n0 * DH))[t];
      ((uint4*)lv)[t] = ((const uint4*)(Vp + (size_t)n0 * DH))[t];
    }
#endif
    __syncthreads();                                // tile visible to all waves

#pragma unroll
    for (int kk2 = 0; kk2 < 64; kk2 += 32) {
      Frag a;                                       // A[d][n] = k'[n][d] gather (LDS)
#pragma unroll
      for (int i = 0; i < 16; ++i) {
        const int kn = kk2 + ((i < 8) ? i + 8 * half : i + 8 + 8 * half);
        a.u[i] = lk[kn * DH + m];
      }
      Frag bb[2];
#pragma unroll
      for (int j = 0; j < 2; ++j) {
        for (int i = 0; i < 16; ++i)
          bb[j].u[i] = lv[(kk2 + ((i < 8) ? i + 8 * half : i + 8 + 8 * half)) * DH + (tn0 + j) * 16 + ln];
      }
#pragma unroll
      for (int j = 0; j < 2; ++j)
        acc[j] = __builtin_amdgcn_wmma_f32_16x16x32_bf16(
            false, a.v, false, bb[j].v, (short)0, acc[j], false, false);
    }
  }

  unsigned short* cp = CTX + (size_t)bh * DH * DH;
  for (int j = 0; j < 2; ++j) {
    const int e = (tn0 + j) * 16 + ln;
    for (int r = 0; r < 8; ++r) {
      const int d = tm * 16 + r + 8 * half;
      cp[(size_t)d * DH + e] = f2bf(acc[j][r]);
    }
  }
}

// ----------------- kernel 6: out = q * ctx, fused spatial transpose write
// grid (32, 128), block 128 (4 waves). Wave: 32 n-rows x full e=64.
__global__ void out_gemm(const unsigned short* __restrict__ Q,
                         const unsigned short* __restrict__ CTX,
                         unsigned short* __restrict__ Y) {
  const int lane = threadIdx.x & 31, wave = threadIdx.x >> 5;
  const int bh = blockIdx.y;
  const int b = bh >> 3, head = bh & 7;
  const int nbase = blockIdx.x * 128 + wave * 32;
  const int ln = lane & 15, half = lane >> 4;
  const unsigned short* Qp = Q   + (size_t)bh * N * DH;
  const unsigned short* Cp = CTX + (size_t)bh * DH * DH;
  const v8f vz = {0.f,0.f,0.f,0.f,0.f,0.f,0.f,0.f};
  v8f acc[2][4];
  for (int i = 0; i < 2; ++i) for (int j = 0; j < 4; ++j) acc[i][j] = vz;

  for (int kk = 0; kk < DH; kk += 32) {
    Frag a[2];
#pragma unroll
    for (int tm = 0; tm < 2; ++tm) {
      const uint4* p = (const uint4*)(Qp + (size_t)(nbase + tm * 16 + ln) * DH + kk + 8 * half);
      a[tm].q[0] = p[0]; a[tm].q[1] = p[2];
    }
    Frag bb[4];
#pragma unroll
    for (int tn = 0; tn < 4; ++tn) {
      for (int i = 0; i < 16; ++i)
        bb[tn].u[i] = Cp[(size_t)(kk + ((i < 8) ? i + 8 * half : i + 8 + 8 * half)) * DH + tn * 16 + ln];
    }
#pragma unroll
    for (int tm = 0; tm < 2; ++tm)
#pragma unroll
      for (int tn = 0; tn < 4; ++tn)
        acc[tm][tn] = __builtin_amdgcn_wmma_f32_16x16x32_bf16(
            false, a[tm].v, false, bb[tn].v, (short)0, acc[tm][tn], false, false);
  }

  unsigned short* Yb = Y + (size_t)b * HID * N;
  for (int tm = 0; tm < 2; ++tm) {
    const int nb   = nbase + tm * 16 + 8 * half;    // n = nb + r
    const int widx = nb & 63;                       // n%64 (stays in-range, nb%8==0)
    const int hidx = nb >> 6;
    for (int tn = 0; tn < 4; ++tn) {
      const int e = tn * 16 + ln;
      unsigned short* yp = Yb + (size_t)(head * DH + e) * N + (size_t)widx * 64 + hidx;
      for (int r = 0; r < 8; ++r)
        yp[(size_t)r * 64] = f2bf(acc[tm][tn][r]);  // m_pix = (widx+r)*64 + hidx
    }
  }
}

// ------------- kernel 7: final 1x1 conv (256x512 x 512xN) + channel LN
// grid (128, 16), block 256 (8 waves). Block owns full M=256 x N=32.
__global__ void final_gemm_ln(const unsigned short* __restrict__ Wo,
                              const unsigned short* __restrict__ Y,
                              const float* __restrict__ gout,
                              float* __restrict__ out) {
  __shared__ float tile[256 * 32];
  __shared__ float smu[32], srs[32];
  const int lane = threadIdx.x & 31, wave = threadIdx.x >> 5;
  const int b   = blockIdx.y;
  const int nb0 = blockIdx.x * 32;
  const int ln = lane & 15, half = lane >> 4;
  const unsigned short* Yb = Y + (size_t)b * HID * N;
  const v8f vz = {0.f,0.f,0.f,0.f,0.f,0.f,0.f,0.f};
  v8f acc[2][2];
  for (int i = 0; i < 2; ++i) for (int j = 0; j < 2; ++j) acc[i][j] = vz;

  for (int kk = 0; kk < HID; kk += 32) {
    Frag a[2];
#pragma unroll
    for (int tm = 0; tm < 2; ++tm) {
      const uint4* p = (const uint4*)(Wo + (size_t)(wave * 32 + tm * 16 + ln) * HID + kk + 8 * half);
      a[tm].q[0] = p[0]; a[tm].q[1] = p[2];
    }
    Frag bb[2];
#pragma unroll
    for (int tn = 0; tn < 2; ++tn) {
      for (int i = 0; i < 16; ++i)
        bb[tn].u[i] = Yb[(size_t)(kk + ((i < 8) ? i + 8 * half : i + 8 + 8 * half)) * N + nb0 + tn * 16 + ln];
    }
#pragma unroll
    for (int tm = 0; tm < 2; ++tm)
#pragma unroll
      for (int tn = 0; tn < 2; ++tn)
        acc[tm][tn] = __builtin_amdgcn_wmma_f32_16x16x32_bf16(
            false, a[tm].v, false, bb[tn].v, (short)0, acc[tm][tn], false, false);
  }

  for (int tm = 0; tm < 2; ++tm)
    for (int tn = 0; tn < 2; ++tn) {
      const int col = tn * 16 + ln;
      const int ob  = wave * 32 + tm * 16 + 8 * half;
      for (int r = 0; r < 8; ++r)
        tile[(ob + r) * 32 + col] = acc[tm][tn][r];
    }
  __syncthreads();
  if (threadIdx.x < 32) {
    const int n = threadIdx.x;
    float s = 0.f, ss = 0.f;
    for (int o = 0; o < 256; ++o) { float v = tile[o * 32 + n]; s += v; ss += v * v; }
    const float mu = s * (1.0f / 256.0f);
    smu[n] = mu;
    srs[n] = rsqrtf(ss * (1.0f / 256.0f) - mu * mu + 1e-5f);
  }
  __syncthreads();
  const int n  = threadIdx.x & 31;
  const int ob = threadIdx.x >> 5;
  for (int j = 0; j < 32; ++j) {
    const int o = ob + 8 * j;
    const float v = (tile[o * 32 + n] - smu[n]) * srs[n] * gout[o];
    out[(size_t)(b * 256 + o) * N + nb0 + n] = v;
  }
}

// ---------------------------------------------------------------- launch
extern "C" void kernel_launch(void* const* d_in, const int* in_sizes, int n_in,
                              void* d_out, int out_size, void* d_ws, size_t ws_size,
                              hipStream_t stream) {
  const float* feat  = (const float*)d_in[0];
  const float* g_in  = (const float*)d_in[1];
  const float* wqkv  = (const float*)d_in[2];
  const float* wout  = (const float*)d_in[3];
  const float* g_out = (const float*)d_in[4];
  float* out = (float*)d_out;

  char* ws = (char*)d_ws;                      // ~228 MB used
  unsigned short* X   = (unsigned short*)(ws + 0);          //  32 MiB (B,C,N) bf16
  unsigned short* WQ  = (unsigned short*)(ws + 33554432);   // 768 KiB
  unsigned short* WO  = (unsigned short*)(ws + 34340864);   // 256 KiB
  unsigned short* Q   = (unsigned short*)(ws + 34603008);   //  64 MiB (BH,N,64)
  unsigned short* Kb  = (unsigned short*)(ws + 101711872);  //  64 MiB
  unsigned short* V   = (unsigned short*)(ws + 168820736);  //  64 MiB
  unsigned short* CTX = (unsigned short*)(ws + 235929600);  //   2 MiB
  float2*         ST  = (float2*)(ws + 238026752);          //  64 KiB
  unsigned short* Y   = V;                                  // reuse V after ctx_gemm

  cvt_f32_bf16<<<dim3((K3 * C + 255) / 256), 256, 0, stream>>>(wqkv, WQ, K3 * C);
  cvt_f32_bf16<<<dim3((C * HID + 255) / 256), 256, 0, stream>>>(wout, WO, C * HID);
  ln_in<<<dim3(BATCH * N / 256), 256, 0, stream>>>(feat, g_in, X);
  qkv_gemm<<<dim3(32, 24, 16), 128, 0, stream>>>(WQ, X, Q, Kb, V);
  k_colstats<<<dim3(BH), 256, 0, stream>>>(Kb, ST);
  k_norm<<<dim3(2048), 256, 0, stream>>>(Kb, ST);
  ctx_gemm<<<dim3(BH), 256, 0, stream>>>(Kb, V, CTX);
  out_gemm<<<dim3(32, BH), 128, 0, stream>>>(Q, CTX, Y);
  final_gemm_ln<<<dim3(128, BATCH), 256, 0, stream>>>(WO, Y, g_out, out);
}
